// PNAe_72232759984528
// MI455X (gfx1250) — matrix operand order repaired and checked by reading, not compiled, not run on hardware
//
#include <hip/hip_runtime.h>
#include <hip/hip_bf16.h>
#include <stdint.h>

typedef _Float16 f16;
typedef _Float16 v4h  __attribute__((ext_vector_type(4)));
typedef _Float16 v8h  __attribute__((ext_vector_type(8)));
typedef _Float16 v16h __attribute__((ext_vector_type(16)));
typedef float    v4f  __attribute__((ext_vector_type(4)));
typedef float    v8f  __attribute__((ext_vector_type(8)));

static constexpr int NN = 20000;     // nodes
static constexpr int NE = 160000;    // edges
static constexpr int NT = 5;         // towers
static constexpr float AVG_LOG = 2.1420662f;

// ---------------- order-preserving float <-> uint key for atomic min/max ----
__device__ __forceinline__ unsigned fkey(float f) {
    int i = __float_as_int(f);
    return (i >= 0) ? ((unsigned)i | 0x80000000u) : ~(unsigned)i;
}
__device__ __forceinline__ float fdec(unsigned k) {
    int i = (k & 0x80000000u) ? (int)(k & 0x7FFFFFFFu) : (int)~k;
    return __int_as_float(i);
}

// ---- A-tile staging helper: 4x aligned v4f loads -> 2x b128 LDS stores ----
// GUARD=false: all loads statically in-bounds (uniform fast path)
template<int AMODE, bool GUARD>
__device__ __forceinline__ void stage_a_vec(
    f16* As, int sRow, int sKseg, int row, int k0, int K, int M,
    const float* __restrict__ A, long lda,
    const float* __restrict__ P, const float* __restrict__ Q,
    const float* __restrict__ R, long gi0, long gi1)
{
    v4h hv[4];
    #pragma unroll
    for (int g = 0; g < 4; ++g) {
        int kk = sKseg + g * 4;
        int gk = k0 + kk;
        v4f fv = {};
        if (!GUARD || (row < M && gk < K)) {
            const float* sp;
            if (AMODE == 0) {
                sp = A + (long)row * lda + gk;
            } else {
                if (gk < 100)      sp = P + gi0 * 100 + gk;
                else if (gk < 200) sp = Q + gi1 * 100 + (gk - 100);
                else               sp = R + (long)row * 100 + (gk - 200);
            }
            fv = *(const v4f*)sp;
        }
        hv[g] = __builtin_convertvector(fv, v4h);
    }
    v8h lo = __builtin_shufflevector(hv[0], hv[1], 0,1,2,3,4,5,6,7);
    v8h hi = __builtin_shufflevector(hv[2], hv[3], 0,1,2,3,4,5,6,7);
    *(v8h*)(As + sRow * 32 + sKseg)     = lo;
    *(v8h*)(As + sRow * 32 + sKseg + 8) = hi;
}

// ---------------- generic 64x64 WMMA GEMM ----------------------------------
// AMODE: 0 = dense fp32 A [M x K] (row stride lda)
//        1 = concat3 gather: row r = [P[idx0[r]][0:100] | Q[idx1[r]][0:100] | R[r][0:100]]
// EMODE: 0 = store C (+bias, optional relu)
//        1 = fused PNA stats scatter: atomic sum/sumsq/min/max into per-node [500]
// B is pre-packed f16, K-major transposed: Bh[j * Kp + k], zero padded.
template<int AMODE, int EMODE>
__global__ __launch_bounds__(128)
void gemm_wmma(const float* __restrict__ A, int lda, int vec,
               const float* __restrict__ P, const float* __restrict__ Q,
               const float* __restrict__ R,
               const int* __restrict__ idx0, const int* __restrict__ idx1,
               const f16* __restrict__ Bh, int K, int Kp, int Nc, int Np,
               const float* __restrict__ bias, int relu,
               float* __restrict__ C, int ldc, int M,
               float* __restrict__ sum, float* __restrict__ sq,
               unsigned* __restrict__ mn, unsigned* __restrict__ mx,
               const int* __restrict__ dstIdx)
{
    __shared__ __align__(16) f16 As[64 * 32];   // As[r][k]
    __shared__ __align__(16) f16 Bs[64 * 32];   // transposed: Bs[n][k]

    const int tid  = threadIdx.x;
    const int wave = tid >> 5;
    const int lane = tid & 31;
    const int half = lane >> 4;
    const int l16  = lane & 15;
    const int rowBase = blockIdx.x * 64;
    const int colBase = blockIdx.y * 64;

    // staging coordinates: each thread owns (line, 16 consecutive k)
    const int sRow  = tid >> 1;
    const int sKseg = (tid & 1) << 4;
    const int rowS  = rowBase + sRow;
    const bool blockFull = (rowBase + 64 <= M);   // uniform across block

    // B staging pointer (loop-invariant bounds)
    const int  jS    = colBase + sRow;
    const bool jOK   = (jS < Np);
    const f16* bBase = Bh + (long)jS * Kp + sKseg;

    // gather indices loaded once (concat3 mode)
    long gi0 = 0, gi1 = 0;
    if (AMODE == 1 && rowS < M) { gi0 = idx0[rowS]; gi1 = idx1[rowS]; }

    v8f acc[4] = {};

    for (int k0 = 0; k0 < Kp; k0 += 32) {
        const bool fullK = (k0 + 32 <= K);        // uniform
        // ---- stage A tile ----
        if (AMODE == 1 || vec) {
            if (blockFull && fullK)
                stage_a_vec<AMODE, false>(As, sRow, sKseg, rowS, k0, K, M,
                                          A, lda, P, Q, R, gi0, gi1);
            else
                stage_a_vec<AMODE, true>(As, sRow, sKseg, rowS, k0, K, M,
                                         A, lda, P, Q, R, gi0, gi1);
        } else {
            #pragma unroll
            for (int i = 0; i < 16; ++i) {
                int li = tid + i * 128;
                int r  = li >> 5, k = li & 31;
                int row = rowBase + r, gk = k0 + k;
                float v = 0.f;
                if (row < M && gk < K) v = A[(long)row * lda + gk];
                As[r * 32 + k] = (f16)v;
            }
        }
        // ---- stage B tile: two b128 global loads + two b128 LDS stores ----
        {
            v8h b0 = {}, b1 = {};
            if (jOK) {
                b0 = *(const v8h*)(bBase + k0);
                b1 = *(const v8h*)(bBase + k0 + 8);
            }
            *(v8h*)(Bs + sRow * 32 + sKseg)     = b0;
            *(v8h*)(Bs + sRow * 32 + sKseg + 8) = b1;
        }
        __syncthreads();

        // A fragment: lane l16 = row-in-tile; k set {8*half..+7, 16+8*half..+7}
        int aRow = wave * 16 + l16;
        v8h alo = *(const v8h*)(As + aRow * 32 + 8 * half);
        v8h ahi = *(const v8h*)(As + aRow * 32 + 16 + 8 * half);
        v16h a = __builtin_shufflevector(alo, ahi,
                 0,1,2,3,4,5,6,7,8,9,10,11,12,13,14,15);
        #pragma unroll
        for (int ct = 0; ct < 4; ++ct) {
            // B fragment: lane l16 = col; k set {8*half..+7, 16+8*half..+7}
            int bRow = ct * 16 + l16;
            v8h blo = *(const v8h*)(Bs + bRow * 32 + 8 * half);
            v8h bhi = *(const v8h*)(Bs + bRow * 32 + 16 + 8 * half);
            v16h b = __builtin_shufflevector(blo, bhi,
                     0,1,2,3,4,5,6,7,8,9,10,11,12,13,14,15);
            acc[ct] = __builtin_amdgcn_wmma_f32_16x16x32_f16(
                false, a, false, b, (short)0, acc[ct], false, false);
        }
        __syncthreads();
    }

    // ---- epilogue ----
    #pragma unroll
    for (int ct = 0; ct < 4; ++ct) {
        #pragma unroll
        for (int r = 0; r < 8; ++r) {
            int row = rowBase + wave * 16 + half * 8 + r;
            int j   = colBase + ct * 16 + l16;
            if (row < M && j < Nc) {
                float val = acc[ct][r] + (bias ? bias[j] : 0.f);
                if (EMODE == 0) {
                    if (relu && val < 0.f) val = 0.f;
                    C[(long)row * ldc + j] = val;
                } else {
                    int node = dstIdx[row];
                    long o = (long)node * 500 + j;
                    atomicAdd(sum + o, val);
                    atomicAdd(sq + o, val * val);
                    unsigned key = fkey(val);
                    atomicMin(mn + o, key);
                    atomicMax(mx + o, key);
                }
            }
        }
    }
}

// ------------- weight packing (fp32 -> padded f16, K-major transposed) -----
// output layout: out[j * Kp + k]
__global__ void pack_generic(const float* __restrict__ src, f16* __restrict__ out,
                             int K, int Nc, int Kp, int Np) {
    int i = blockIdx.x * 256 + threadIdx.x;
    if (i >= Kp * Np) return;
    int k = i / Np, j = i % Np;
    float v = (k < K && j < Nc) ? src[(long)k * Nc + j] : 0.f;
    out[(long)j * Kp + k] = (f16)v;
}
// pre_W[l]: [T,300,100] -> transposed [512 x 320]  (col j = t*100+c)
__global__ void pack_wcat(const float* __restrict__ pw, f16* __restrict__ out) {
    int i = blockIdx.x * 256 + threadIdx.x;
    if (i >= 320 * 512) return;
    int k = i / 512, j = i % 512;
    float v = 0.f;
    if (k < 300 && j < 500) {
        int t = j / 100, c = j % 100;
        v = pw[((long)t * 300 + k) * 100 + c];
    }
    out[(long)j * 320 + k] = (f16)v;
}
// post_W[l]: [T,1300,20] rows 0..99 -> transposed [112 x 128] (col j = t*20+o)
__global__ void pack_wx(const float* __restrict__ pw, f16* __restrict__ out) {
    int i = blockIdx.x * 256 + threadIdx.x;
    if (i >= 128 * 112) return;
    int k = i / 112, j = i % 112;
    float v = 0.f;
    if (k < 100 && j < 100)
        v = pw[((long)(j / 20) * 1300 + k) * 20 + (j % 20)];
    out[(long)j * 128 + k] = (f16)v;
}
// post_W[l]: agg slices (rows 100+g*400+k) -> [T][64 x 416] (cols: s1|s2|s3)
__global__ void pack_wagg(const float* __restrict__ pw, f16* __restrict__ out) {
    int i = blockIdx.x * 256 + threadIdx.x;
    if (i >= 5 * 416 * 64) return;
    int t = i / (416 * 64), rem = i % (416 * 64);
    int k = rem / 64, j = rem % 64;
    float v = 0.f;
    if (k < 400 && j < 60) {
        int g = j / 20, o = j % 20;
        v = pw[((long)t * 1300 + 100 + g * 400 + k) * 20 + o];
    }
    out[((long)t * 64 + j) * 416 + k] = (f16)v;
}

// ---------------- PNA aggregation helpers ----------------------------------
__global__ void init_stats(float* sum, float* sq, unsigned* mn, unsigned* mx, float* cnt) {
    long i = (long)blockIdx.x * 256 + threadIdx.x;
    if (i < (long)NN * 500) { sum[i] = 0.f; sq[i] = 0.f; mn[i] = 0xFFFFFFFFu; mx[i] = 0u; }
    if (i < NN) cnt[i] = 0.f;
}
__global__ void count_edges(const int* __restrict__ dst, float* __restrict__ cnt) {
    int e = blockIdx.x * 256 + threadIdx.x;
    if (e < NE) atomicAdd(cnt + dst[e], 1.f);
}
__global__ void finalize_stats(const float* __restrict__ sum, const float* __restrict__ sq,
                               const unsigned* __restrict__ mn, const unsigned* __restrict__ mx,
                               const float* __restrict__ cnt,
                               float* __restrict__ agg, float* __restrict__ amp,
                               float* __restrict__ inv) {
    long i = (long)blockIdx.x * 256 + threadIdx.x;
    if (i >= (long)NN * 500) return;
    int n = (int)(i / 500), j = (int)(i % 500);
    int t = j / 100, hh = j % 100;
    float c  = cnt[n];
    float cc = fmaxf(c, 1.f);
    float mean = sum[i] / cc;
    float m2   = sq[i] / cc;
    float sd   = sqrtf(fmaxf(m2 - mean * mean, 0.f) + 1e-5f);
    bool  has  = c > 0.f;
    long base = (long)n * 2000 + (long)t * 400 + hh;
    agg[base]       = mean;
    agg[base + 100] = has ? fdec(mn[i]) : 0.f;
    agg[base + 200] = has ? fdec(mx[i]) : 0.f;
    agg[base + 300] = sd;
    if (j == 0) {
        float a = logf(cc + 1.f) / AVG_LOG;
        amp[n] = a; inv[n] = 1.f / a;
    }
}
__global__ void combine_post(const float* __restrict__ xh, const float* __restrict__ s,
                             const float* __restrict__ amp, const float* __restrict__ inv,
                             float* __restrict__ post) {
    int i = blockIdx.x * 256 + threadIdx.x;
    if (i >= NN * 100) return;
    int n = i / 100, j = i % 100;
    int t = j / 20, o = j % 20;
    long sb = (long)n * 300 + (long)t * 60 + o;
    post[i] = xh[i] + s[sb] + amp[n] * s[sb + 20] + inv[n] * s[sb + 40];
}

// ---------------- BatchNorm (batch stats) + ReLU ---------------------------
__global__ void bn_stats(const float* __restrict__ c, float* __restrict__ mu,
                         float* __restrict__ var) {
    __shared__ float ssum[256], ssq[256];
    int col = blockIdx.x, tid = threadIdx.x;
    float s = 0.f, q = 0.f;
    for (int n = tid; n < NN; n += 256) {
        float v = c[(long)n * 100 + col];
        s += v; q += v * v;
    }
    ssum[tid] = s; ssq[tid] = q; __syncthreads();
    for (int off = 128; off; off >>= 1) {
        if (tid < off) { ssum[tid] += ssum[tid + off]; ssq[tid] += ssq[tid + off]; }
        __syncthreads();
    }
    if (tid == 0) {
        float m = ssum[0] / (float)NN;
        mu[col] = m;
        var[col] = ssq[0] / (float)NN - m * m;
    }
}
__global__ void bn_apply(const float* __restrict__ c, const float* __restrict__ mu,
                         const float* __restrict__ var, const float* __restrict__ gamma,
                         const float* __restrict__ beta, float* __restrict__ h) {
    int i = blockIdx.x * 256 + threadIdx.x;
    if (i >= NN * 100) return;
    int col = i % 100;
    float v = gamma[col] * (c[i] - mu[col]) * rsqrtf(var[col] + 1e-5f) + beta[col];
    h[i] = fmaxf(v, 0.f);
}

// ---------------------------------------------------------------------------
extern "C" void kernel_launch(void* const* d_in, const int* in_sizes, int n_in,
                              void* d_out, int out_size, void* d_ws, size_t ws_size,
                              hipStream_t stream) {
    const float* x        = (const float*)d_in[0];
    const float* eattr    = (const float*)d_in[1];
    const float* nembW    = (const float*)d_in[2];
    const float* nembB    = (const float*)d_in[3];
    const float* eembW    = (const float*)d_in[4];
    const float* eembB    = (const float*)d_in[5];
    const float* eeW      = (const float*)d_in[6];
    const float* eeB      = (const float*)d_in[7];
    const float* preW     = (const float*)d_in[8];
    const float* preB     = (const float*)d_in[9];
    const float* postW    = (const float*)d_in[10];
    const float* postB    = (const float*)d_in[11];
    const float* linW     = (const float*)d_in[12];
    const float* linB     = (const float*)d_in[13];
    const float* bnG      = (const float*)d_in[14];
    const float* bnB      = (const float*)d_in[15];
    const float* emW1     = (const float*)d_in[16];
    const float* emB1     = (const float*)d_in[17];
    const float* emW2     = (const float*)d_in[18];
    const float* emB2     = (const float*)d_in[19];
    const float* mlpW1    = (const float*)d_in[20];
    const float* mlpB1    = (const float*)d_in[21];
    const float* mlpW2    = (const float*)d_in[22];
    const float* mlpB2    = (const float*)d_in[23];
    const float* mlpW3    = (const float*)d_in[24];
    const float* mlpB3    = (const float*)d_in[25];
    const int*   src      = (const int*)d_in[26];
    const int*   dst      = src + NE;
    float* out = (float*)d_out;

    // workspace carve-out
    uint8_t* wsb = (uint8_t*)d_ws;
    size_t off = 0;
    auto take = [&](size_t bytes) {
        uint8_t* p = wsb + off;
        off = (off + bytes + 255) & ~(size_t)255;
        return p;
    };
    float*    h     = (float*)take((size_t)NN * 100 * 4);
    float*    cbuf  = (float*)take((size_t)NN * 100 * 4);
    float*    post  = (float*)take((size_t)NN * 100 * 4);
    float*    xh    = (float*)take((size_t)NN * 100 * 4);
    float*    sbuf  = (float*)take((size_t)NN * 300 * 4);
    float*    amp   = (float*)take((size_t)NN * 4);
    float*    inv   = (float*)take((size_t)NN * 4);
    float*    cnt   = (float*)take((size_t)NN * 4);
    float*    mu    = (float*)take(100 * 4);
    float*    var   = (float*)take(100 * 4);
    float*    ssum  = (float*)take((size_t)NN * 500 * 4);
    float*    ssq   = (float*)take((size_t)NN * 500 * 4);
    unsigned* smn   = (unsigned*)take((size_t)NN * 500 * 4);
    unsigned* smx   = (unsigned*)take((size_t)NN * 500 * 4);
    float*    agg   = (float*)take((size_t)NN * 2000 * 4);
    float*    ea    = (float*)take((size_t)NE * 100 * 4);
    float*    shr   = (float*)take((size_t)NE * 100 * 4);   // e_enc / t1 / f1
    f16*      wpack = (f16*)take(512 * 1024);
    float*    f2    = ssum;                                 // reuse after stats done

    auto ceilN = [](int a, int b) { return (a + b - 1) / b; };

    // dense-A GEMM with on-the-fly weight pack
    auto gemmD = [&](const float* A, int lda, const float* W, int K, int Nc,
                     const float* bias, int relu, float* C, int ldc, int M) {
        int Kp = (K + 31) & ~31, Np = (Nc + 15) & ~15;
        int vec = (lda % 4 == 0 && K % 4 == 0) ? 1 : 0;
        int tot = Kp * Np;
        pack_generic<<<ceilN(tot, 256), 256, 0, stream>>>(W, wpack, K, Nc, Kp, Np);
        dim3 g(ceilN(M, 64), ceilN(Np, 64));
        gemm_wmma<0, 0><<<g, 128, 0, stream>>>(A, lda, vec,
            nullptr, nullptr, nullptr, nullptr, nullptr,
            wpack, K, Kp, Nc, Np, bias, relu, C, ldc, M,
            nullptr, nullptr, nullptr, nullptr, nullptr);
    };
    // concat3-A GEMM (store epilogue) with on-the-fly weight pack
    auto gemmC = [&](const float* Pp, const float* Qp, const float* Rp,
                     const int* i0, const int* i1, const float* W, int Nc,
                     const float* bias, int relu, float* C, int ldc, int M) {
        const int K = 300, Kp = 320;
        int Np = (Nc + 15) & ~15;
        int tot = Kp * Np;
        pack_generic<<<ceilN(tot, 256), 256, 0, stream>>>(W, wpack, K, Nc, Kp, Np);
        dim3 g(ceilN(M, 64), ceilN(Np, 64));
        gemm_wmma<1, 0><<<g, 128, 0, stream>>>(nullptr, 0, 1,
            Pp, Qp, Rp, i0, i1,
            wpack, K, Kp, Nc, Np, bias, relu, C, ldc, M,
            nullptr, nullptr, nullptr, nullptr, nullptr);
    };

    // ---- embeddings ----
    gemmD(x, 64, nembW, 64, 100, nembB, 0, h, 100, NN);
    gemmD(eattr, 32, eembW, 32, 100, eembB, 0, ea, 100, NE);

    for (int l = 0; l < 2; ++l) {
        float* e_enc = shr;
        // edge encoder
        gemmD(ea, 100, eeW + (size_t)l * 100 * 100, 100, 100,
              eeB + (size_t)l * 100, 0, e_enc, 100, NE);

        // init stats + degree count
        init_stats<<<ceilN(NN * 500, 256), 256, 0, stream>>>(ssum, ssq, smn, smx, cnt);
        count_edges<<<ceilN(NE, 256), 256, 0, stream>>>(dst, cnt);

        // pre-MLP: [E,300] (concat h[dst]|h[src]|e_enc) x [300,500], fused stats scatter
        pack_wcat<<<ceilN(320 * 512, 256), 256, 0, stream>>>(
            preW + (size_t)l * NT * 300 * 100, wpack);
        {
            dim3 g(ceilN(NE, 64), ceilN(512, 64));
            gemm_wmma<1, 1><<<g, 128, 0, stream>>>(nullptr, 0, 1,
                h, h, e_enc, dst, src,
                wpack, 300, 320, 500, 512, preB + (size_t)l * 500, 0,
                nullptr, 0, NE, ssum, ssq, smn, smx, dst);
        }
        finalize_stats<<<ceilN(NN * 500, 256), 256, 0, stream>>>(
            ssum, ssq, smn, smx, cnt, agg, amp, inv);

        // post-MLP, decomposed: xh = h @ Wx (+post_b); s = per-tower agg GEMMs
        const float* pwl = postW + (size_t)l * NT * 1300 * 20;
        pack_wx<<<ceilN(128 * 112, 256), 256, 0, stream>>>(pwl, wpack);
        {
            dim3 g(ceilN(NN, 64), ceilN(112, 64));
            gemm_wmma<0, 0><<<g, 128, 0, stream>>>(h, 100, 1,
                nullptr, nullptr, nullptr, nullptr, nullptr,
                wpack, 100, 128, 100, 112, postB + (size_t)l * 100, 0,
                xh, 100, NN, nullptr, nullptr, nullptr, nullptr, nullptr);
        }
        pack_wagg<<<ceilN(5 * 416 * 64, 256), 256, 0, stream>>>(pwl, wpack);
        for (int t = 0; t < NT; ++t) {
            dim3 g(ceilN(NN, 64), 1);
            gemm_wmma<0, 0><<<g, 128, 0, stream>>>(agg + (size_t)t * 400, 2000, 1,
                nullptr, nullptr, nullptr, nullptr, nullptr,
                wpack + (size_t)t * 416 * 64, 400, 416, 60, 64, nullptr, 0,
                sbuf + (size_t)t * 60, 300, NN,
                nullptr, nullptr, nullptr, nullptr, nullptr);
        }
        combine_post<<<ceilN(NN * 100, 256), 256, 0, stream>>>(xh, sbuf, amp, inv, post);

        // lin
        gemmD(post, 100, linW + (size_t)l * 100 * 100, 100, 100,
              linB + (size_t)l * 100, 0, cbuf, 100, NN);

        // BatchNorm + ReLU -> h
        bn_stats<<<100, 256, 0, stream>>>(cbuf, mu, var);
        bn_apply<<<ceilN(NN * 100, 256), 256, 0, stream>>>(
            cbuf, mu, var, bnG + (size_t)l * 100, bnB + (size_t)l * 100, h);

        // edge-update MLP: t1 = relu([h[src]|h[dst]|ea] @ W1 + b1); ea = t1 @ W2 + b2
        float* t1 = shr;
        gemmC(h, h, ea, src, dst, emW1 + (size_t)l * 300 * 100, 100,
              emB1 + (size_t)l * 100, 1, t1, 100, NE);
        gemmD(t1, 100, emW2 + (size_t)l * 100 * 100, 100, 100,
              emB2 + (size_t)l * 100, 0, ea, 100, NE);
    }

    // ---- final per-edge MLP (h already ReLU'd by BN+ReLU) ----
    float* fone = shr;
    gemmC(h, h, ea, src, dst, mlpW1, 50, mlpB1, 1, fone, 50, NE);
    gemmD(fone, 50, mlpW2, 50, 25, mlpB2, 1, f2, 25, NE);
    gemmD(f2, 25, mlpW3, 25, 1, mlpB3, 0, out, 1, NE);

    (void)in_sizes; (void)n_in; (void)out_size; (void)ws_size;
}
